// DAGNNConv_51505247814282
// MI455X (gfx1250) — compile-verified, matrix operand order checked by reading, not run on hardware
//
#include <hip/hip_runtime.h>
#include <cstdint>
#include <cstddef>

typedef __attribute__((ext_vector_type(16))) _Float16 v16h;
typedef __attribute__((ext_vector_type(8)))  float    v8f;

#define CFEAT 1024
#undef CFEAT
#define C 256
#define NHOP 6   // K+1 planes (x plus 5 propagated)

// ---------------- GCN norm precompute ----------------

__global__ void init_deg_kernel(float* __restrict__ deg, int n) {
    int i = blockIdx.x * blockDim.x + threadIdx.x;
    if (i < n) deg[i] = 1.0f;   // self-loop contribution
}

__global__ void count_deg_kernel(const long long* __restrict__ ei, int m,
                                 float* __restrict__ deg,
                                 int* __restrict__ row32, int* __restrict__ col32) {
    int e = blockIdx.x * blockDim.x + threadIdx.x;
    if (e < m) {
        int r = (int)ei[e];
        int c = (int)ei[(size_t)m + e];
        row32[e] = r;
        col32[e] = c;
        unsafeAtomicAdd(&deg[c], 1.0f);
    }
}

__global__ void calc_dis_kernel(const float* __restrict__ deg, float* __restrict__ dis, int n) {
    int i = blockIdx.x * blockDim.x + threadIdx.x;
    if (i < n) {
        float d = deg[i];
        dis[i] = (d > 0.0f) ? rsqrtf(d) : 0.0f;
    }
}

__global__ void calc_norm_kernel(const int* __restrict__ row32, const int* __restrict__ col32,
                                 const float* __restrict__ dis, float* __restrict__ enorm, int m) {
    int e = blockIdx.x * blockDim.x + threadIdx.x;
    if (e < m) enorm[e] = dis[row32[e]] * dis[col32[e]];
}

// ---------------- SpMM pass ----------------

// hout = dis[i]^2 * hin   (self-loop edge, also initializes hout)
__global__ void self_loop_kernel(const float* __restrict__ hin, float* __restrict__ hout,
                                 const float* __restrict__ dis, int n) {
    int idx = blockIdx.x * blockDim.x + threadIdx.x;   // one float4 per thread
    if (idx < n * (C / 4)) {
        int node = idx >> 6;          // C/4 == 64
        int q    = idx & 63;
        float s = dis[node]; s = s * s;
        float4 v = *((const float4*)(hin + (size_t)node * C) + q);
        float4 o = make_float4(s * v.x, s * v.y, s * v.z, s * v.w);
        *((float4*)(hout + (size_t)node * C) + q) = o;
    }
}

// hout[row] += norm * hin[col]; 4 edges/block, 64 threads/edge, float4 per thread
__global__ void spmm_scatter_kernel(const float* __restrict__ hin, float* __restrict__ hout,
                                    const int* __restrict__ row32, const int* __restrict__ col32,
                                    const float* __restrict__ enorm, int m) {
    int t = threadIdx.x;
    int e = blockIdx.x * 4 + (t >> 6);
    if (e < m) {
        int r = row32[e];
        int c = col32[e];
        float w = enorm[e];
        int q = t & 63;
        float4 v = *((const float4*)(hin + (size_t)c * C) + q);
        float* dst = hout + (size_t)r * C + (size_t)q * 4;
        unsafeAtomicAdd(dst + 0, w * v.x);
        unsafeAtomicAdd(dst + 1, w * v.y);
        unsafeAtomicAdd(dst + 2, w * v.z);
        unsafeAtomicAdd(dst + 3, w * v.w);
    }
}

// ---------------- DAGNN combine (WMMA) ----------------

__device__ __forceinline__ float sigmoidf_(float x) {
    return 1.0f / (1.0f + __expf(-x));
}

// One wave per 16-node tile; 8 waves (tiles) per 256-thread block.
// Phase 1: dots[k][m] = H_k[m,:] . w  via v_wmma_f32_16x16x32_f16 chained over 8 K-chunks.
// Phase 2: out[m,:] = sum_k sigmoid(dots+b) * H_k[m,:], float4-vectorized.
__global__ void dagnn_combine_kernel(const float* __restrict__ x, const float* __restrict__ H,
                                     const float* __restrict__ w, const float* __restrict__ bptr,
                                     float* __restrict__ out, int n) {
    __shared__ float dots[8][NHOP][16];

    const int lane = threadIdx.x & 31;
    const int wid  = threadIdx.x >> 5;
    const int tile = blockIdx.x * 8 + wid;
    const int numTiles = (n + 15) >> 4;
    const size_t NC = (size_t)n * C;

    const float* hp[NHOP];
    hp[0] = x;
#pragma unroll
    for (int k = 1; k < NHOP; ++k) hp[k] = H + (size_t)(k - 1) * NC;
    const float bias = bptr[0];

    if (tile < numTiles) {
        const int m = lane & 15;        // row of tile this lane contributes in A-fragment
        const int g = lane >> 4;        // lane group selects K sub-ranges
        int nodeA = tile * 16 + m;
        if (nodeA >= n) nodeA = n - 1;  // clamp pad rows (results discarded)
        const int k0off = g ? 8  : 0;
        const int k1off = g ? 24 : 16;

        v8f acc[NHOP];
#pragma unroll
        for (int k = 0; k < NHOP; ++k) acc[k] = (v8f){0, 0, 0, 0, 0, 0, 0, 0};

        for (int ch = 0; ch < 8; ++ch) {           // 8 chunks of K=32 cover C=256
            const int base = ch * 32;
            // B fragment: w replicated into all 16 columns
            v16h bf;
#pragma unroll
            for (int j = 0; j < 8; ++j) {
                bf[j]     = (_Float16)w[base + k0off + j];
                bf[8 + j] = (_Float16)w[base + k1off + j];
            }
#pragma unroll
            for (int k = 0; k < NHOP; ++k) {
                const float* hr = hp[k] + (size_t)nodeA * C + base;
                v16h af;
#pragma unroll
                for (int j = 0; j < 8; ++j) {
                    af[j]     = (_Float16)hr[k0off + j];
                    af[8 + j] = (_Float16)hr[k1off + j];
                }
                acc[k] = __builtin_amdgcn_wmma_f32_16x16x32_f16(
                    false, af, false, bf, (short)0, acc[k], false, false);
            }
        }
        // C/D layout: element r of lane-group g holds row M = g*8 + r (all 16 cols equal)
        if ((lane & 15) == 0) {
#pragma unroll
            for (int k = 0; k < NHOP; ++k)
#pragma unroll
                for (int r = 0; r < 8; ++r)
                    dots[wid][k][g * 8 + r] = acc[k][r];
        }
    }
    __syncthreads();

    if (tile < numTiles) {
        const int f1 = lane * 4;   // lanes cover features [0,128) and [128,256) as float4s
        for (int m = 0; m < 16; ++m) {
            int node = tile * 16 + m;
            if (node >= n) break;  // lane-uniform
            float s[NHOP];
#pragma unroll
            for (int k = 0; k < NHOP; ++k)
                s[k] = sigmoidf_(dots[wid][k][m] + bias);
            float4 a1 = make_float4(0, 0, 0, 0);
            float4 a2 = make_float4(0, 0, 0, 0);
#pragma unroll
            for (int k = 0; k < NHOP; ++k) {
                const float* hr = hp[k] + (size_t)node * C;
                float4 v1 = *((const float4*)(hr + f1));
                float4 v2 = *((const float4*)(hr + 128 + f1));
                a1.x += s[k] * v1.x; a1.y += s[k] * v1.y; a1.z += s[k] * v1.z; a1.w += s[k] * v1.w;
                a2.x += s[k] * v2.x; a2.y += s[k] * v2.y; a2.z += s[k] * v2.z; a2.w += s[k] * v2.w;
            }
            *((float4*)(out + (size_t)node * C + f1))       = a1;
            *((float4*)(out + (size_t)node * C + 128 + f1)) = a2;
        }
    }
}

// ---------------- launch ----------------

extern "C" void kernel_launch(void* const* d_in, const int* in_sizes, int n_in,
                              void* d_out, int out_size, void* d_ws, size_t ws_size,
                              hipStream_t stream) {
    const float*     x  = (const float*)d_in[0];
    const long long* ei = (const long long*)d_in[1];
    const float*     lw = (const float*)d_in[2];
    const float*     lb = (const float*)d_in[3];
    float*           outp = (float*)d_out;

    const int N = in_sizes[0] / C;
    const int M = in_sizes[1] / 2;

    // workspace layout
    float* deg   = (float*)d_ws;
    float* dis   = deg + N;
    int*   row32 = (int*)(dis + N);
    int*   col32 = row32 + M;
    float* enorm = (float*)(col32 + M);
    size_t off   = (size_t)2 * N + (size_t)3 * M;
    off          = (off + 3) & ~(size_t)3;          // 16-byte align H planes
    float* H     = (float*)d_ws + off;              // 5 planes of N*C floats
    const size_t NC = (size_t)N * C;

    const int T = 256;

    // GCN normalization
    init_deg_kernel<<<(N + T - 1) / T, T, 0, stream>>>(deg, N);
    count_deg_kernel<<<(M + T - 1) / T, T, 0, stream>>>(ei, M, deg, row32, col32);
    calc_dis_kernel<<<(N + T - 1) / T, T, 0, stream>>>(deg, dis, N);
    calc_norm_kernel<<<(M + T - 1) / T, T, 0, stream>>>(row32, col32, dis, enorm, M);

    // K=5 propagation passes (self-loop fused into per-pass init)
    const float* hin = x;
    for (int k = 0; k < 5; ++k) {
        float* hout = H + (size_t)k * NC;
        self_loop_kernel<<<(N * (C / 4) + T - 1) / T, T, 0, stream>>>(hin, hout, dis, N);
        spmm_scatter_kernel<<<(M + 3) / 4, T, 0, stream>>>(hin, hout, row32, col32, enorm, M);
        hin = hout;
    }

    // adaptive combine with WMMA dot products
    const int numTiles = (N + 15) / 16;
    dagnn_combine_kernel<<<(numTiles + 7) / 8, T, 0, stream>>>(x, H, lw, lb, outp, N);
}
